// PathGNN_34703335752375
// MI455X (gfx1250) — compile-verified, matrix operand order checked by reading, not run on hardware
//
#include <hip/hip_runtime.h>
#include <hip/hip_bf16.h>
#include <math.h>

typedef __bf16 bf16;
typedef __attribute__((ext_vector_type(16))) __bf16 v16bf;
typedef __attribute__((ext_vector_type(8)))  __bf16 v8bf;
typedef __attribute__((ext_vector_type(8)))  float  v8f;

#define PP 512
#define TT 256
#define HH 256
#define EE 256
#define FF 64
#define G4 1024   // 4*H

// ---------------- WMMA helpers (CDNA5 16x16x32 bf16, f32 accumulate) -------

__device__ __forceinline__ v8f zero8() {
  v8f z;
#pragma unroll
  for (int i = 0; i < 8; ++i) z[i] = 0.f;
  return z;
}

// A/B fragment from row-major storage. Caller pre-adjusts `p` to
// rowBase + (lane&15)*stride + (lane>>4)*8; elements [0..7] at p, [8..15] at p+16.
__device__ __forceinline__ v16bf frag_ld(const bf16* p) {
  v8bf lo = *(const v8bf*)(p);
  v8bf hi = *(const v8bf*)(p + 16);
  v16bf r;
#pragma unroll
  for (int i = 0; i < 8; ++i) { r[i] = lo[i]; r[i + 8] = hi[i]; }
  return r;
}

// Same fragment gathered from f32 memory with on-the-fly bf16 convert.
__device__ __forceinline__ v16bf frag_ld_f32(const float* p) {
  v16bf r;
#pragma unroll
  for (int i = 0; i < 8; ++i) { r[i] = (__bf16)p[i]; r[i + 8] = (__bf16)p[i + 16]; }
  return r;
}

__device__ __forceinline__ v8f wmma_bf16(v16bf a, v16bf b, v8f c) {
  // (neg_a, A, neg_b, B, c_mod, C, reuse_a, reuse_b)
  return __builtin_amdgcn_wmma_f32_16x16x32_bf16(false, a, false, b, (short)0, c,
                                                 false, false);
}

__device__ __forceinline__ float sigm(float x) { return 1.f / (1.f + __expf(-x)); }

// ---------------- utility kernels ------------------------------------------

__global__ void k_f32_to_bf16(const float* __restrict__ s, bf16* __restrict__ d, int n) {
  int i = blockIdx.x * blockDim.x + threadIdx.x;
  if (i < n) d[i] = (__bf16)s[i];
}
__global__ void k_zero_f32(float* d, int n) {
  int i = blockIdx.x * blockDim.x + threadIdx.x;
  if (i < n) d[i] = 0.f;
}
__global__ void k_zero_bf16(bf16* d, int n) {
  int i = blockIdx.x * blockDim.x + threadIdx.x;
  if (i < n) d[i] = (__bf16)0.f;
}

// ---------------- gather-embed + projection: X = emb[seq] @ Wp^T + bp ------
// M = P*T (gathered rows), N = H, K = E. 4 waves/block; each wave owns two
// 16-row tiles (32 rows) x 64 cols. Double-buffered fragment loads.

__global__ void __launch_bounds__(128) k_embed(
    const int* __restrict__ seq,
    const float* __restrict__ emb,   // [N_nodes, E] f32
    const bf16* __restrict__ WpB,    // [H, E] bf16 row-major
    const float* __restrict__ bp,    // [H]
    bf16* __restrict__ X)            // [P*T, H] bf16
{
  const int lane = threadIdx.x & 31;
  const int wv   = threadIdx.x >> 5;            // wave 0..3 -> 32-row strip
  const int rowBase = blockIdx.x * 128 + wv * 32;
  const int colBase = blockIdx.y * 64;
  const int r  = lane & 15;
  const int kh = (lane >> 4) * 8;

  const float* arow[2];
  arow[0] = emb + (long)seq[rowBase + r] * EE + kh;
  arow[1] = emb + (long)seq[rowBase + 16 + r] * EE + kh;

  const bf16* brow[4];
#pragma unroll
  for (int c = 0; c < 4; ++c)
    brow[c] = WpB + (long)(colBase + c * 16 + r) * EE + kh;

  v8f acc[2][4];
#pragma unroll
  for (int rt = 0; rt < 2; ++rt)
#pragma unroll
    for (int c = 0; c < 4; ++c) acc[rt][c] = zero8();

  // prologue: load k=0 fragments
  v16bf ca[2], cb[4];
#pragma unroll
  for (int rt = 0; rt < 2; ++rt) ca[rt] = frag_ld_f32(arow[rt]);
#pragma unroll
  for (int c = 0; c < 4; ++c) cb[c] = frag_ld(brow[c]);

#pragma unroll 1
  for (int k = 0; k < EE - 32; k += 32) {
    v16bf na[2], nb[4];
#pragma unroll
    for (int rt = 0; rt < 2; ++rt) na[rt] = frag_ld_f32(arow[rt] + k + 32);
#pragma unroll
    for (int c = 0; c < 4; ++c) nb[c] = frag_ld(brow[c] + k + 32);
#pragma unroll
    for (int rt = 0; rt < 2; ++rt)
#pragma unroll
      for (int c = 0; c < 4; ++c)
        acc[rt][c] = wmma_bf16(ca[rt], cb[c], acc[rt][c]);
#pragma unroll
    for (int rt = 0; rt < 2; ++rt) ca[rt] = na[rt];
#pragma unroll
    for (int c = 0; c < 4; ++c) cb[c] = nb[c];
  }
#pragma unroll
  for (int rt = 0; rt < 2; ++rt)
#pragma unroll
    for (int c = 0; c < 4; ++c)
      acc[rt][c] = wmma_bf16(ca[rt], cb[c], acc[rt][c]);

  const int n0 = colBase + (lane & 15);
#pragma unroll
  for (int rt = 0; rt < 2; ++rt) {
    const int mrow = rowBase + rt * 16 + (lane >> 4) * 8;
#pragma unroll
    for (int c = 0; c < 4; ++c) {
      const int n  = n0 + c * 16;
      const float bn = bp[n];
#pragma unroll
      for (int v = 0; v < 8; ++v)
        X[(long)(mrow + v) * HH + n] = (__bf16)(acc[rt][c][v] + bn);
    }
  }
}

// ---------------- fused LSTM step ------------------------------------------
// One launch per timestep, blockIdx.z = direction (fwd/rev run concurrently).
// Each wave owns two 16-row tiles (32 rows) x 32 hidden cols and computes
// ALL FOUR gate tiles -> 16 WMMAs per 32-k step off 10 fragment loads, with
// explicit double-buffered loads so WMMAs overlap the next step's loads.

struct DirArgs {
  const bf16*  x;      // sequence input base
  const bf16*  Wi;     // [4H, D] bf16
  const bf16*  Wh;     // [4H, H] bf16
  const float* bias;   // [4H] f32
  const bf16*  hIn;    // [P, H] bf16 (read)
  bf16*        hOut;   // [P, H] bf16 (write; ping-pong vs hIn)
  float*       hF;     // [P, H] f32 state
  float*       cF;     // [P, H] f32 state
  bf16*        oOut;   // layer-0 output (O0) or nullptr
  long         xOff;   // element offset of this timestep in x
  long         oOff;   // element col offset (incl. time) in O0
  int          tIdx;   // actual time index for masking
};

struct StepArgs {
  DirArgs d[2];
  const int* lengths;
  long xStride;        // row stride of x (elements)
  long oStride;        // row stride of O0 (elements)
};

template <int KD>    // K of the Wi GEMM: 256 (layer0) or 512 (layer1)
__global__ void __launch_bounds__(128) k_lstm_step(StepArgs A)
{
  const DirArgs dd = A.d[blockIdx.z];
  const int lane = threadIdx.x & 31;
  const int wv   = threadIdx.x >> 5;            // 4 waves -> 32-row strips
  const int rowBase = blockIdx.x * 128 + wv * 32;
  const int colBase = blockIdx.y * 32;          // 32 hidden cols per block
  const int r  = lane & 15;
  const int kh = (lane >> 4) * 8;

  v8f acc[2][4][2];
#pragma unroll
  for (int rt = 0; rt < 2; ++rt)
#pragma unroll
    for (int g = 0; g < 4; ++g)
#pragma unroll
      for (int c = 0; c < 2; ++c) acc[rt][g][c] = zero8();

  // ---- segment 1: z += x_t @ Wi^T  (K = KD), double-buffered
  {
    const bf16* a0 = dd.x + (long)(rowBase + r) * A.xStride + dd.xOff + kh;
    const bf16* a1 = a0 + 16 * A.xStride;
    const bf16* bw[4][2];
#pragma unroll
    for (int g = 0; g < 4; ++g)
#pragma unroll
      for (int c = 0; c < 2; ++c)
        bw[g][c] = dd.Wi + (long)(g * HH + colBase + c * 16 + r) * KD + kh;

    v16bf ca[2], cb[4][2];
    ca[0] = frag_ld(a0); ca[1] = frag_ld(a1);
#pragma unroll
    for (int g = 0; g < 4; ++g)
#pragma unroll
      for (int c = 0; c < 2; ++c) cb[g][c] = frag_ld(bw[g][c]);

#pragma unroll 1
    for (int k = 0; k < KD - 32; k += 32) {
      v16bf na[2], nb[4][2];
      na[0] = frag_ld(a0 + k + 32);
      na[1] = frag_ld(a1 + k + 32);
#pragma unroll
      for (int g = 0; g < 4; ++g)
#pragma unroll
        for (int c = 0; c < 2; ++c) nb[g][c] = frag_ld(bw[g][c] + k + 32);
#pragma unroll
      for (int rt = 0; rt < 2; ++rt)
#pragma unroll
        for (int g = 0; g < 4; ++g)
#pragma unroll
          for (int c = 0; c < 2; ++c)
            acc[rt][g][c] = wmma_bf16(ca[rt], cb[g][c], acc[rt][g][c]);
      ca[0] = na[0]; ca[1] = na[1];
#pragma unroll
      for (int g = 0; g < 4; ++g)
#pragma unroll
        for (int c = 0; c < 2; ++c) cb[g][c] = nb[g][c];
    }
#pragma unroll
    for (int rt = 0; rt < 2; ++rt)
#pragma unroll
      for (int g = 0; g < 4; ++g)
#pragma unroll
        for (int c = 0; c < 2; ++c)
          acc[rt][g][c] = wmma_bf16(ca[rt], cb[g][c], acc[rt][g][c]);
  }

  // ---- segment 2: z += h_{t-1} @ Wh^T  (K = H), double-buffered
  {
    const bf16* a0 = dd.hIn + (long)(rowBase + r) * HH + kh;
    const bf16* a1 = a0 + 16 * HH;
    const bf16* bw[4][2];
#pragma unroll
    for (int g = 0; g < 4; ++g)
#pragma unroll
      for (int c = 0; c < 2; ++c)
        bw[g][c] = dd.Wh + (long)(g * HH + colBase + c * 16 + r) * HH + kh;

    v16bf ca[2], cb[4][2];
    ca[0] = frag_ld(a0); ca[1] = frag_ld(a1);
#pragma unroll
    for (int g = 0; g < 4; ++g)
#pragma unroll
      for (int c = 0; c < 2; ++c) cb[g][c] = frag_ld(bw[g][c]);

#pragma unroll 1
    for (int k = 0; k < HH - 32; k += 32) {
      v16bf na[2], nb[4][2];
      na[0] = frag_ld(a0 + k + 32);
      na[1] = frag_ld(a1 + k + 32);
#pragma unroll
      for (int g = 0; g < 4; ++g)
#pragma unroll
        for (int c = 0; c < 2; ++c) nb[g][c] = frag_ld(bw[g][c] + k + 32);
#pragma unroll
      for (int rt = 0; rt < 2; ++rt)
#pragma unroll
        for (int g = 0; g < 4; ++g)
#pragma unroll
          for (int c = 0; c < 2; ++c)
            acc[rt][g][c] = wmma_bf16(ca[rt], cb[g][c], acc[rt][g][c]);
      ca[0] = na[0]; ca[1] = na[1];
#pragma unroll
      for (int g = 0; g < 4; ++g)
#pragma unroll
        for (int c = 0; c < 2; ++c) cb[g][c] = nb[g][c];
    }
#pragma unroll
    for (int rt = 0; rt < 2; ++rt)
#pragma unroll
      for (int g = 0; g < 4; ++g)
#pragma unroll
        for (int c = 0; c < 2; ++c)
          acc[rt][g][c] = wmma_bf16(ca[rt], cb[g][c], acc[rt][g][c]);
  }

  // ---- LSTM cell + length masking (f32, matches reference semantics)
  const int n0 = colBase + (lane & 15);
#pragma unroll
  for (int rt = 0; rt < 2; ++rt) {
    const int mrow = rowBase + rt * 16 + (lane >> 4) * 8;
#pragma unroll
    for (int c = 0; c < 2; ++c) {
      const int n = n0 + c * 16;
      const float bi  = dd.bias[0 * HH + n];
      const float bf_ = dd.bias[1 * HH + n];
      const float bg  = dd.bias[2 * HH + n];
      const float bo  = dd.bias[3 * HH + n];
#pragma unroll
      for (int v = 0; v < 8; ++v) {
        const int  m   = mrow + v;
        const long idx = (long)m * HH + n;
        const float zi = acc[rt][0][c][v] + bi;
        const float zf = acc[rt][1][c][v] + bf_;
        const float zg = acc[rt][2][c][v] + bg;
        const float zo = acc[rt][3][c][v] + bo;
        const float cold = dd.cF[idx];
        const float hold = dd.hF[idx];
        const float cn = sigm(zf) * cold + sigm(zi) * tanhf(zg);
        const float hn = sigm(zo) * tanhf(cn);
        const bool valid = dd.tIdx < A.lengths[m];
        const float hv = valid ? hn : hold;
        const float cv = valid ? cn : cold;
        dd.hF[idx]   = hv;
        dd.cF[idx]   = cv;
        dd.hOut[idx] = (__bf16)hv;
        if (dd.oOut)
          dd.oOut[(long)m * A.oStride + dd.oOff + n] = (__bf16)hv;
      }
    }
  }
}

// ---------------- tiny f32 epilogue (<0.3 GFLOP total) ----------------------

__global__ void k_pathfeat(const float* __restrict__ features, const float* __restrict__ Wf,
                           const float* __restrict__ bfv, float* __restrict__ pf) {
  int i = blockIdx.x * blockDim.x + threadIdx.x;     // i = p*H + j
  if (i >= PP * HH) return;
  int p = i / HH, j = i % HH;
  float s = bfv[j];
  const float* fr = features + p * FF;
  const float* wr = Wf + j * FF;
#pragma unroll 8
  for (int e = 0; e < FF; ++e) s += fr[e] * wr[e];
  pf[i] = s;
}

__global__ void k_fuse(const float* __restrict__ h2, const float* __restrict__ h3,
                       const float* __restrict__ pf, const float* __restrict__ Wfus,
                       const float* __restrict__ bfus, float* __restrict__ pemb) {
  int i = blockIdx.x * blockDim.x + threadIdx.x;     // i = p*H + j
  if (i >= PP * HH) return;
  int p = i / HH, j = i % HH;
  const float* w = Wfus + (long)j * 3 * HH;
  const float* a = h2 + p * HH;
  const float* b = h3 + p * HH;
  const float* c = pf + p * HH;
  float s = bfus[j];
  for (int k = 0; k < HH; ++k) s += a[k] * w[k];
  for (int k = 0; k < HH; ++k) s += b[k] * w[HH + k];
  for (int k = 0; k < HH; ++k) s += c[k] * w[2 * HH + k];
  pemb[i] = fmaxf(s, 0.f);
}

__global__ void k_attn(const float* __restrict__ pemb, const float* __restrict__ Wa,
                       const float* __restrict__ ba, float* __restrict__ logits) {
  int p = blockIdx.x * blockDim.x + threadIdx.x;
  if (p >= PP) return;
  float s = ba[0];
  const float* e = pemb + p * HH;
  for (int k = 0; k < HH; ++k) s += e[k] * Wa[k];
  logits[p] = s;
}

__global__ void k_pool(const float* __restrict__ pemb, const float* __restrict__ logits,
                       float* __restrict__ agg) {
  __shared__ float sm[256];
  const int j = threadIdx.x;                         // 256 threads, 1 block
  float lm = fmaxf(logits[j], logits[j + 256]);
  sm[j] = lm; __syncthreads();
  for (int s = 128; s > 0; s >>= 1) { if (j < s) sm[j] = fmaxf(sm[j], sm[j + s]); __syncthreads(); }
  const float mx = sm[0]; __syncthreads();
  float le = __expf(logits[j] - mx) + __expf(logits[j + 256] - mx);
  sm[j] = le; __syncthreads();
  for (int s = 128; s > 0; s >>= 1) { if (j < s) sm[j] += sm[j + s]; __syncthreads(); }
  const float Z = sm[0];
  float accW = 0.f, accS = 0.f, accM = -3.402823466e38f;
  for (int p = 0; p < PP; ++p) {
    const float e = pemb[p * HH + j];
    accW += e * __expf(logits[p] - mx);
    accS += e;
    accM = fmaxf(accM, e);
  }
  agg[j]          = accW / Z;
  agg[HH + j]     = accS / (float)PP;
  agg[2 * HH + j] = accM;
}

__global__ void k_mlp1(const float* __restrict__ agg, const float* __restrict__ W1,
                       const float* __restrict__ b1, float* __restrict__ hmid) {
  int j = blockIdx.x * blockDim.x + threadIdx.x;
  if (j >= 2 * HH) return;
  const float* w = W1 + (long)j * 3 * HH;
  float s = b1[j];
  for (int k = 0; k < 3 * HH; ++k) s += agg[k] * w[k];
  hmid[j] = fmaxf(s, 0.f);
}

__global__ void k_mlp2(const float* __restrict__ hmid, const float* __restrict__ W2,
                       const float* __restrict__ b2, float* __restrict__ out) {
  int j = blockIdx.x * blockDim.x + threadIdx.x;
  if (j >= HH) return;
  const float* w = W2 + (long)j * 2 * HH;
  float s = b2[j];
  for (int k = 0; k < 2 * HH; ++k) s += hmid[k] * w[k];
  out[j] = s;
}

// ---------------- host launcher --------------------------------------------

extern "C" void kernel_launch(void* const* d_in, const int* in_sizes, int n_in,
                              void* d_out, int out_size, void* d_ws, size_t ws_size,
                              hipStream_t stream) {
  (void)in_sizes; (void)n_in; (void)out_size; (void)ws_size;

  const int*   seq      = (const int*)d_in[0];
  const int*   lengths  = (const int*)d_in[1];
  const float* features = (const float*)d_in[2];
  const float* emb      = (const float*)d_in[3];
  const float* Wp   = (const float*)d_in[5];
  const float* bp   = (const float*)d_in[6];
  const float* Wi0f = (const float*)d_in[7];
  const float* Wh0f = (const float*)d_in[8];
  const float* b0f  = (const float*)d_in[9];
  const float* Wi0r = (const float*)d_in[10];
  const float* Wh0r = (const float*)d_in[11];
  const float* b0r  = (const float*)d_in[12];
  const float* Wi1f = (const float*)d_in[13];
  const float* Wh1f = (const float*)d_in[14];
  const float* b1f  = (const float*)d_in[15];
  const float* Wi1r = (const float*)d_in[16];
  const float* Wh1r = (const float*)d_in[17];
  const float* b1r  = (const float*)d_in[18];
  const float* Wf   = (const float*)d_in[19];
  const float* bfv  = (const float*)d_in[20];
  const float* Wfus = (const float*)d_in[21];
  const float* bfus = (const float*)d_in[22];
  const float* Wa   = (const float*)d_in[23];
  const float* ba   = (const float*)d_in[24];
  const float* W1   = (const float*)d_in[25];
  const float* b1v  = (const float*)d_in[26];
  const float* W2   = (const float*)d_in[27];
  const float* b2v  = (const float*)d_in[28];

  // ---- carve workspace (deterministic layout every call)
  char* wptr = (char*)d_ws;
  auto carve = [&](size_t bytes) -> char* {
    char* p = wptr;
    wptr += (bytes + 255) & ~(size_t)255;
    return p;
  };
  bf16*  X     = (bf16*)carve((size_t)PP * TT * HH * 2);       // 64 MB
  bf16*  O0    = (bf16*)carve((size_t)PP * TT * 2 * HH * 2);   // 128 MB
  bf16*  WpB   = (bf16*)carve((size_t)HH * EE * 2);
  bf16*  Wi0fB = (bf16*)carve((size_t)G4 * HH * 2);
  bf16*  Wh0fB = (bf16*)carve((size_t)G4 * HH * 2);
  bf16*  Wi0rB = (bf16*)carve((size_t)G4 * HH * 2);
  bf16*  Wh0rB = (bf16*)carve((size_t)G4 * HH * 2);
  bf16*  Wi1fB = (bf16*)carve((size_t)G4 * 2 * HH * 2);
  bf16*  Wh1fB = (bf16*)carve((size_t)G4 * HH * 2);
  bf16*  Wi1rB = (bf16*)carve((size_t)G4 * 2 * HH * 2);
  bf16*  Wh1rB = (bf16*)carve((size_t)G4 * HH * 2);
  bf16*  hPing = (bf16*)carve((size_t)4 * 2 * PP * HH * 2);    // [dirlayer][buf][P*H]
  float* hF    = (float*)carve((size_t)4 * PP * HH * 4);
  float* cF    = (float*)carve((size_t)4 * PP * HH * 4);
  float* pf    = (float*)carve((size_t)PP * HH * 4);
  float* pemb  = (float*)carve((size_t)PP * HH * 4);
  float* logits= (float*)carve(512 * 4);
  float* agg   = (float*)carve(768 * 4);
  float* hmid  = (float*)carve(512 * 4);

  // ---- weight conversion f32 -> bf16 (small, once per launch)
  auto cvt = [&](const float* s, bf16* d, int n) {
    k_f32_to_bf16<<<(n + 255) / 256, 256, 0, stream>>>(s, d, n);
  };
  cvt(Wp,   WpB,   HH * EE);
  cvt(Wi0f, Wi0fB, G4 * HH);     cvt(Wh0f, Wh0fB, G4 * HH);
  cvt(Wi0r, Wi0rB, G4 * HH);     cvt(Wh0r, Wh0rB, G4 * HH);
  cvt(Wi1f, Wi1fB, G4 * 2 * HH); cvt(Wh1f, Wh1fB, G4 * HH);
  cvt(Wi1r, Wi1rB, G4 * 2 * HH); cvt(Wh1r, Wh1rB, G4 * HH);

  // ---- zero initial states
  {
    int nb = 4 * 2 * PP * HH;
    k_zero_bf16<<<(nb + 255) / 256, 256, 0, stream>>>(hPing, nb);
    int nf = 4 * PP * HH;
    k_zero_f32<<<(nf + 255) / 256, 256, 0, stream>>>(hF, nf);
    k_zero_f32<<<(nf + 255) / 256, 256, 0, stream>>>(cF, nf);
  }

  // ---- gather-embed + projection (WMMA)
  k_embed<<<dim3((PP * TT) / 128, HH / 64), 128, 0, stream>>>(seq, emb, WpB, bp, X);

  const size_t PH = (size_t)PP * HH;
  const dim3 stepGrid(PP / 128, HH / 32, 2);

  // ---- layer 0 (fwd + rev fused per launch; writes O0 = [o0f | o0r])
  for (int t = 0; t < TT; ++t) {
    StepArgs S;
    S.lengths = lengths;
    S.xStride = (long)TT * HH; S.oStride = (long)TT * 2 * HH;
    const int tr = TT - 1 - t;
    // fwd (dir-layer 0)
    S.d[0].x = X;  S.d[0].Wi = Wi0fB; S.d[0].Wh = Wh0fB; S.d[0].bias = b0f;
    S.d[0].hIn  = hPing + (0 * 2 + (t & 1)) * PH;
    S.d[0].hOut = hPing + (0 * 2 + ((t + 1) & 1)) * PH;
    S.d[0].hF = hF + 0 * PH; S.d[0].cF = cF + 0 * PH;
    S.d[0].oOut = O0; S.d[0].xOff = (long)t * HH;
    S.d[0].oOff = (long)t * 2 * HH; S.d[0].tIdx = t;
    // rev (dir-layer 1)
    S.d[1].x = X;  S.d[1].Wi = Wi0rB; S.d[1].Wh = Wh0rB; S.d[1].bias = b0r;
    S.d[1].hIn  = hPing + (1 * 2 + (t & 1)) * PH;
    S.d[1].hOut = hPing + (1 * 2 + ((t + 1) & 1)) * PH;
    S.d[1].hF = hF + 1 * PH; S.d[1].cF = cF + 1 * PH;
    S.d[1].oOut = O0; S.d[1].xOff = (long)tr * HH;
    S.d[1].oOff = (long)tr * 2 * HH + HH; S.d[1].tIdx = tr;
    k_lstm_step<HH><<<stepGrid, 128, 0, stream>>>(S);
  }

  // ---- layer 1 (input = O0, D = 2H; only final hidden needed)
  for (int t = 0; t < TT; ++t) {
    StepArgs S;
    S.lengths = lengths;
    S.xStride = (long)TT * 2 * HH; S.oStride = 0;
    const int tr = TT - 1 - t;
    S.d[0].x = O0; S.d[0].Wi = Wi1fB; S.d[0].Wh = Wh1fB; S.d[0].bias = b1f;
    S.d[0].hIn  = hPing + (2 * 2 + (t & 1)) * PH;
    S.d[0].hOut = hPing + (2 * 2 + ((t + 1) & 1)) * PH;
    S.d[0].hF = hF + 2 * PH; S.d[0].cF = cF + 2 * PH;
    S.d[0].oOut = nullptr; S.d[0].xOff = (long)t * 2 * HH;
    S.d[0].oOff = 0; S.d[0].tIdx = t;
    S.d[1].x = O0; S.d[1].Wi = Wi1rB; S.d[1].Wh = Wh1rB; S.d[1].bias = b1r;
    S.d[1].hIn  = hPing + (3 * 2 + (t & 1)) * PH;
    S.d[1].hOut = hPing + (3 * 2 + ((t + 1) & 1)) * PH;
    S.d[1].hF = hF + 3 * PH; S.d[1].cF = cF + 3 * PH;
    S.d[1].oOut = nullptr; S.d[1].xOff = (long)tr * 2 * HH;
    S.d[1].oOff = 0; S.d[1].tIdx = tr;
    k_lstm_step<2 * HH><<<stepGrid, 128, 0, stream>>>(S);
  }

  // ---- epilogue: feature proj, fusion MLP, attention pool, head
  k_pathfeat<<<(PP * HH + 255) / 256, 256, 0, stream>>>(features, Wf, bfv, pf);
  k_fuse<<<(PP * HH + 255) / 256, 256, 0, stream>>>(hF + 2 * PH, hF + 3 * PH, pf,
                                                    Wfus, bfus, pemb);
  k_attn<<<(PP + 255) / 256, 256, 0, stream>>>(pemb, Wa, ba, logits);
  k_pool<<<1, 256, 0, stream>>>(pemb, logits, agg);
  k_mlp1<<<(2 * HH + 255) / 256, 256, 0, stream>>>(agg, W1, b1v, hmid);
  k_mlp2<<<(HH + 255) / 256, 256, 0, stream>>>(hmid, W2, b2v, (float*)d_out);
}
